// IntensityLoss_4346506903553
// MI455X (gfx1250) — matrix operand order repaired and checked by reading, not compiled, not run on hardware
//
#include <hip/hip_runtime.h>
#include <hip/hip_bf16.h>

// CTLSTM intensity loss for MI455X (gfx1250, wave32).
// Memory-bound: 256MB of f32 row data @ 23.3TB/s ~ 11us floor; trans pipe
// (2 x 184M exp/tanh) is the matching compute floor. Each wave owns 16 batch
// rows; row data (cx-cbarx, cbarx, deltx*(-log2e) folded, ox) is loaded ONCE
// per 32-wide h-chunk and reused across all 11 intensity evaluations.
// H-reduction (h_t . W) via v_wmma_f32_16x16x32_f16: A = W broadcast to all
// 16 rows (LDS-staged), B = h_t^T, 11 f32 accumulators chained over 8 chunks.
// exp(-d*t) is computed as exp2(d * tn[e]) with tn[e] = -log2(e)*t_e
// precomputed per evaluation -> single v_mul + v_exp_f32 per element.

typedef __attribute__((ext_vector_type(16))) _Float16 v16h;
typedef __attribute__((ext_vector_type(8)))  float    v8f;

#define H_DIM  256
#define NCHUNK 8      // H / 32
#define NS     10
#define NEV    11     // 1 (at duration) + NS samples
#define EPS_F  1e-8f
#define LOG2E_F 1.4426950408889634f

static __device__ __forceinline__ float fast_exp2(float x) {
#if __has_builtin(__builtin_amdgcn_exp2f)
  return __builtin_amdgcn_exp2f(x);       // v_exp_f32, no log2e pre-mul
#else
  return exp2f(x);
#endif
}

static __device__ __forceinline__ float fast_tanh(float x) {
#if __has_builtin(__builtin_amdgcn_tanhf)
  return __builtin_amdgcn_tanhf(x);       // v_tanh_f32
#elif __has_builtin(__builtin_amdgcn_tanh_f32)
  return __builtin_amdgcn_tanh_f32(x);
#else
  float e = __expf(2.0f * x);
  return 1.0f - 2.0f / (e + 1.0f);
#endif
}

static __device__ __forceinline__ float softplus_f(float x) {
  // numerically stable: max(x,0) + log(1 + exp(-|x|)); arg of log in (1,2]
  float e = __expf(-fabsf(x));
  return fmaxf(x, 0.0f) + __logf(1.0f + e);
}

__global__ __launch_bounds__(256) void intensity_loss_wmma(
    const float* __restrict__ duration,   // [B]
    const float* __restrict__ cx,         // [B,H]
    const float* __restrict__ cbarx,      // [B,H]
    const float* __restrict__ deltx,      // [B,H]
    const float* __restrict__ ox,         // [B,H]
    const float* __restrict__ W,          // [1,H]
    const float* __restrict__ bvec,       // [1]
    const float* __restrict__ u,          // [NS,B]
    float* __restrict__ out,              // [B]
    int Bn)
{
  __shared__ float sW[H_DIM];
  if (threadIdx.x < H_DIM) sW[threadIdx.x] = W[threadIdx.x];
  __syncthreads();

  const int lane = threadIdx.x & 31;
  const int wave = threadIdx.x >> 5;
  const int n    = lane & 15;       // batch column within 16-row tile (N dim)
  const int hi   = lane >> 4;       // 0: K-low half, 1: K-high half
  const int rowBase = (blockIdx.x * 8 + wave) * 16;
  const int r = rowBase + n;        // this lane's batch row

  const float bb  = bvec[0];
  const float dur = duration[r];

  // tn[e] = -log2(e) * t_e  so that exp(-d*t_e) == exp2(d * tn[e])
  float tn[NEV];
  tn[0] = -LOG2E_F * dur;
#pragma unroll
  for (int e = 0; e < NS; ++e)
    tn[e + 1] = (-LOG2E_F * dur) * u[(size_t)e * (size_t)Bn + r];

  // one f32 16x16 accumulator per evaluation
  v8f acc[NEV];
#pragma unroll
  for (int e = 0; e < NEV; ++e) { v8f z = {}; acc[e] = z; }

  const size_t rowOff = (size_t)r * H_DIM;

#pragma unroll 1
  for (int c = 0; c < NCHUNK; ++c) {
    const int cbase = c * 32;

    // --- A fragment: W replicated into all 16 M-rows ---------------------
    // A 16x32 layout (matches B): lanes 0-15 VGPR j -> K=2j,2j+1 ;
    // lanes 16-31: same + 16.
    v16h aw;
#pragma unroll
    for (int j = 0; j < 8; ++j) {
      int k = cbase + 2 * j + hi * 16;
      float2 w2 = *(const float2*)(sW + k);   // ds_load_b64
      aw[2 * j]     = (_Float16)w2.x;
      aw[2 * j + 1] = (_Float16)w2.y;
    }

    // --- row data: 16 consecutive h-values per lane, loaded once ----------
    // B 32x16 layout: lanes 0-15 hold K=0..15 (this lane's 16 elements in
    // order), lanes 16-31 hold K=16..31 -> natural contiguous float4 loads.
    const size_t off = rowOff + (size_t)cbase + (size_t)hi * 16;
    const float4* cx4 = (const float4*)(cx    + off);
    const float4* cb4 = (const float4*)(cbarx + off);
    const float4* dl4 = (const float4*)(deltx + off);
    const float4* ox4 = (const float4*)(ox    + off);

    float dcxv[16], cbv[16], dlv[16], oxv[16];
#pragma unroll
    for (int q = 0; q < 4; ++q) {
      float4 X  = cx4[q];
      float4 Cb = cb4[q];
      float4 Dl = dl4[q];
      float4 Ox = ox4[q];
      cbv[4*q+0] = Cb.x; cbv[4*q+1] = Cb.y; cbv[4*q+2] = Cb.z; cbv[4*q+3] = Cb.w;
      dcxv[4*q+0] = X.x - Cb.x; dcxv[4*q+1] = X.y - Cb.y;
      dcxv[4*q+2] = X.z - Cb.z; dcxv[4*q+3] = X.w - Cb.w;
      dlv[4*q+0] = Dl.x; dlv[4*q+1] = Dl.y; dlv[4*q+2] = Dl.z; dlv[4*q+3] = Dl.w;
      oxv[4*q+0] = Ox.x; oxv[4*q+1] = Ox.y; oxv[4*q+2] = Ox.z; oxv[4*q+3] = Ox.w;
    }

    // prefetch next chunk of the streaming arrays (global_prefetch_b8)
    if (c + 1 < NCHUNK) {
      __builtin_prefetch(cx    + off + 32, 0, 3);
      __builtin_prefetch(cbarx + off + 32, 0, 3);
      __builtin_prefetch(deltx + off + 32, 0, 3);
      __builtin_prefetch(ox    + off + 32, 0, 3);
    }

    // --- 11 evaluations over this K-chunk ---------------------------------
#pragma unroll
    for (int e = 0; e < NEV; ++e) {
      const float te = tn[e];
      v16h bf;
#pragma unroll
      for (int i = 0; i < 16; ++i) {
        float dec = fast_exp2(dlv[i] * te);              // v_mul + v_exp_f32
        float ct  = __builtin_fmaf(dcxv[i], dec, cbv[i]);
        float ht  = oxv[i] * fast_tanh(ct);              // v_tanh_f32
        bf[i] = (_Float16)ht;
      }
      acc[e] = __builtin_amdgcn_wmma_f32_16x16x32_f16(
          false, aw, false, bf, (short)0, acc[e], false, false);
    }
  }

  // D VGPR0: lanes 0-15 -> M=0 (W row), lanes 16-31 -> M=8 (same W row);
  // every lane holds the dot product for its own batch row r.
  float lam0 = softplus_f(acc[0][0] + bb);
  float res  = -__logf(lam0 + EPS_F);
  float s = 0.0f;
#pragma unroll
  for (int e = 1; e < NEV; ++e) s += softplus_f(acc[e][0] + bb);
  res += s * (1.0f / (float)NS) * dur;

  if (lane < 16) out[r] = res;
}

extern "C" void kernel_launch(void* const* d_in, const int* in_sizes, int n_in,
                              void* d_out, int out_size, void* d_ws, size_t ws_size,
                              hipStream_t stream) {
  const float* duration = (const float*)d_in[0];
  const float* cx       = (const float*)d_in[1];
  const float* cbarx    = (const float*)d_in[2];
  const float* deltx    = (const float*)d_in[3];
  const float* ox       = (const float*)d_in[4];
  const float* W        = (const float*)d_in[5];
  const float* bvec     = (const float*)d_in[6];
  const float* u        = (const float*)d_in[7];
  float* out = (float*)d_out;

  int Bn = in_sizes[0];                 // 65536
  // 256 threads = 8 waves/block, 16 rows/wave -> 128 rows/block
  int grid = (Bn + 127) / 128;
  hipLaunchKernelGGL(intensity_loss_wmma, dim3(grid), dim3(256), 0, stream,
                     duration, cx, cbarx, deltx, ox, W, bvec, u, out, Bn);
}
